// BINN_convnet_17325898072410
// MI455X (gfx1250) — matrix operand order repaired and checked by reading, not compiled, not run on hardware
//
#include <hip/hip_runtime.h>
#include <hip/hip_bf16.h>

typedef __attribute__((ext_vector_type(16))) __bf16 v16bf;
typedef __attribute__((ext_vector_type(8)))  float  v8f;
typedef int v4i __attribute__((vector_size(16)));   // matches builtin's int-vec4 ptr type

#define BM 128          // rows per tile
#define DD 128          // feature dim
#define A_STRIDE 132    // fp32 a-tile LDS row stride (padded for bank spread)
#define W_STRIDE 136    // bf16 W LDS row stride (padded)
#define NGRID 1024      // persistent grid size (tiles processed grid-stride)

#if defined(__HIP_DEVICE_COMPILE__) && \
    __has_builtin(__builtin_amdgcn_global_load_async_to_lds_b128) && \
    __has_builtin(__builtin_amdgcn_s_wait_asynccnt)
#define HAVE_ASYNC 1
#else
#define HAVE_ASYNC 0
#endif

__device__ __forceinline__ void split_bf16(float f, __bf16& h, __bf16& l) {
    h = (__bf16)f;
    l = (__bf16)(f - (float)h);
}

// Stage one 128x128 fp32 tile into (padded) LDS via GLOBAL_LOAD_ASYNC_TO_LDS_B128
// (ASYNCcnt-tracked); 16 instructions per thread.
__device__ __forceinline__ void stage_a_tile(float* dst, const float* __restrict__ src,
                                             int tid) {
    const float4* s4 = (const float4*)src;
#pragma unroll
    for (int it = 0; it < 16; ++it) {
        const int idx4 = tid + it * 256;       // 4096 float4s, 32 per row
        const int r  = idx4 >> 5;
        const int c4 = (idx4 & 31) << 2;
#if HAVE_ASYNC
        __builtin_amdgcn_global_load_async_to_lds_b128(
            (v4i*)(s4 + idx4), (v4i*)(dst + r * A_STRIDE + c4), 0, 0);
#else
        float4 v = s4[idx4];
        float* d = dst + r * A_STRIDE + c4;
        d[0] = v.x; d[1] = v.y; d[2] = v.z; d[3] = v.w;
#endif
    }
}

__device__ __forceinline__ void wait_async_pending16() {
#if HAVE_ASYNC
    __builtin_amdgcn_s_wait_asynccnt(16);   // 16 newer copies outstanding -> older tile drained
#endif
}
__device__ __forceinline__ void wait_async_all() {
#if HAVE_ASYNC
    __builtin_amdgcn_s_wait_asynccnt(0);
#endif
}

__global__ __launch_bounds__(256, 1)
void binn_fused_gemm_stencil(const float* __restrict__ inp,
                             const float* __restrict__ w,      // [128,3]
                             const float* __restrict__ W_lin,  // [128,128]
                             const float* __restrict__ b_lin,  // [128]
                             float* __restrict__ out,
                             int numTiles)
{
    __shared__ float  a_lds[2][BM * A_STRIDE];   // double-buffered fp32 a tiles
    __shared__ __bf16 w_hi[DD * W_STRIDE];       // B[k][j] = W_lin[j*128+k], row j contiguous in k
    __shared__ __bf16 w_lo[DD * W_STRIDE];

    const int tid = threadIdx.x;

    // Kick off async copy of this block's first tile, then stage W while it flies.
    const int t0 = blockIdx.x;                   // grid sized <= numTiles
    stage_a_tile(&a_lds[0][0], inp + (size_t)t0 * BM * DD, tid);

    for (int idx = tid; idx < DD * DD; idx += 256) {
        const int j = idx >> 7;
        const int k = idx & 127;
        float f = W_lin[idx];
        __bf16 h, l; split_bf16(f, h, l);
        w_hi[j * W_STRIDE + k] = h;
        w_lo[j * W_STRIDE + k] = l;
    }

    const int wave = tid >> 5;
    const int lane = tid & 31;
    const int half = lane >> 4;          // 0: lanes 0-15, 1: lanes 16-31
    const int n    = lane & 15;
    const int m_a  = wave * 16 + n;      // A-frag row (16-bit A 16x32 layout)

    // Hoist tile-invariant stencil weights / bias / circular neighbor indices.
    float w0a[8], w1a[8], w2a[8], ba[8];
    int   jp1a[8], jm1a[8], jm2a[8];
#pragma unroll
    for (int jt = 0; jt < 8; ++jt) {
        const int j = jt * 16 + n;
        jp1a[jt] = (j + 1)   & 127;
        jm1a[jt] = (j + 127) & 127;
        jm2a[jt] = (j + 126) & 127;
        w0a[jt]  = w[3 * j + 0];
        w1a[jt]  = w[3 * j + 1];
        w2a[jt]  = (j == 1) ? w[3] : w[3 * j + 2];   // faithful col-1 special case
        ba[jt]   = b_lin[j];
    }

    int cur = 0;
    for (int t = t0; t < numTiles; t += (int)gridDim.x) {
        // Issue next tile's async copy into the other buffer, then drain current.
        const int tn = t + (int)gridDim.x;
        if (tn < numTiles) {
            stage_a_tile(&a_lds[cur ^ 1][0], inp + (size_t)tn * BM * DD, tid);
            wait_async_pending16();
        } else {
            wait_async_all();
        }
        __syncthreads();                 // current buffer (and W on iter 0) ready

        const float* atile = &a_lds[cur][0];
        const int row0 = t * BM;

        v8f acc[8] = {};

        // Occupancy is LDS-limited (1 block/WGP, 2 waves/SIMD), so spend VGPRs
        // freely: fully unroll so the scheduler pipelines ds_loads across k-steps.
#pragma unroll
        for (int kk = 0; kk < DD; kk += 32) {
            // A fragment: fp32 from LDS -> bf16 hi/lo, per 16-bit A 16x32 layout.
            const int c0 = kk + (half ? 16 : 0);
            const float* ap = atile + m_a * A_STRIDE + c0;
            float4 f0 = *(const float4*)(ap + 0);
            float4 f1 = *(const float4*)(ap + 4);
            float4 f2 = *(const float4*)(ap + 8);
            float4 f3 = *(const float4*)(ap + 12);
            float fa[16] = { f0.x, f0.y, f0.z, f0.w, f1.x, f1.y, f1.z, f1.w,
                             f2.x, f2.y, f2.z, f2.w, f3.x, f3.y, f3.z, f3.w };
            v16bf ahi, alo;
#pragma unroll
            for (int e = 0; e < 16; ++e) {
                __bf16 h, l; split_bf16(fa[e], h, l);
                ahi[e] = h; alo[e] = l;
            }

#pragma unroll
            for (int jt = 0; jt < 8; ++jt) {
                const int j  = jt * 16 + n;           // B 32x16: lanes 0-15 K=0..15, 16-31 K=16..31
                const int k0 = kk + (half ? 16 : 0);
                union { uint4 u[2]; v16bf v; } bh, bl;
                const uint4* ph = (const uint4*)&w_hi[j * W_STRIDE + k0];
                bh.u[0] = ph[0]; bh.u[1] = ph[1];
                const uint4* pl = (const uint4*)&w_lo[j * W_STRIDE + k0];
                bl.u[0] = pl[0]; bl.u[1] = pl[1];

                acc[jt] = __builtin_amdgcn_wmma_f32_16x16x32_bf16(
                    false, ahi, false, bh.v, (short)0, acc[jt], false, false);
                acc[jt] = __builtin_amdgcn_wmma_f32_16x16x32_bf16(
                    false, ahi, false, bl.v, (short)0, acc[jt], false, false);
                acc[jt] = __builtin_amdgcn_wmma_f32_16x16x32_bf16(
                    false, alo, false, bh.v, (short)0, acc[jt], false, false);
            }
        }

        // Epilogue: fuse circular stencil + bias; C layout: VGPR i ->
        // lanes 0-15 (M=i, N=lane), lanes 16-31 (M=8+i, N=lane-16).
#pragma unroll
        for (int jt = 0; jt < 8; ++jt) {
            const int j = jt * 16 + n;
#pragma unroll
            for (int i = 0; i < 8; ++i) {
                const int m = wave * 16 + i + half * 8;
                const float ap1 = atile[m * A_STRIDE + jp1a[jt]];
                const float am1 = atile[m * A_STRIDE + jm1a[jt]];
                const float am2 = atile[m * A_STRIDE + jm2a[jt]];
                const float x   = (w0a[jt] * ap1 - w1a[jt] * am2) * (w2a[jt] * am1);
                out[(size_t)(row0 + m) * DD + j] = acc[jt][i] + x + ba[jt];
            }
        }

        __syncthreads();   // everyone done reading buffer `cur` before it is reused
        cur ^= 1;
    }
}

extern "C" void kernel_launch(void* const* d_in, const int* in_sizes, int n_in,
                              void* d_out, int out_size, void* d_ws, size_t ws_size,
                              hipStream_t stream) {
    // setup_inputs order: inp [B,128] f32, dt [1] f32 (unused), w [128,3] f32,
    //                     W_lin [128,128] f32, b_lin [128] f32
    const float* inp   = (const float*)d_in[0];
    const float* w     = (const float*)d_in[2];
    const float* W_lin = (const float*)d_in[3];
    const float* b_lin = (const float*)d_in[4];
    float* out = (float*)d_out;

    const int rows     = in_sizes[0] / DD;     // B
    const int numTiles = rows / BM;            // 4096 for B=524288
    const int blocks   = numTiles < NGRID ? numTiles : NGRID;

    binn_fused_gemm_stencil<<<blocks, 256, 0, stream>>>(inp, w, W_lin, b_lin, out, numTiles);
}
